// SamplingGCN_13975823582074
// MI455X (gfx1250) — matrix operand order repaired and checked by reading, not compiled, not run on hardware
//
#include <hip/hip_runtime.h>
#include <hip/hip_bf16.h>
#include <stdint.h>

// Problem constants (match reference)
#define NN0 200000
#define NN1 50000
#define NN2 10000
#define IN_DIM 512
#define H_DIM 256
#define OUT_DIM 128

typedef float v8f __attribute__((ext_vector_type(8)));
typedef __bf16 bf16x16 __attribute__((ext_vector_type(16)));

union FragCast { uint4 q[2]; bf16x16 v; };

__device__ __forceinline__ bf16x16 ld_frag(const unsigned* p) {
    FragCast u;
    u.q[0] = ((const uint4*)p)[0];
    u.q[1] = ((const uint4*)p)[1];
    return u.v;
}

// Split two f32 into (hi,lo) bf16 pairs, packed as dwords (low half = first elem).
__device__ __forceinline__ void split2(float x0, float x1, unsigned& hi, unsigned& lo) {
    __bf16 h0 = (__bf16)x0, h1 = (__bf16)x1;
    float r0 = x0 - (float)h0, r1 = x1 - (float)h1;
    __bf16 l0 = (__bf16)r0, l1 = (__bf16)r1;
    hi = (unsigned)__builtin_bit_cast(unsigned short, h0) |
         ((unsigned)__builtin_bit_cast(unsigned short, h1) << 16);
    lo = (unsigned)__builtin_bit_cast(unsigned short, l0) |
         ((unsigned)__builtin_bit_cast(unsigned short, l1) << 16);
}

// Error-compensated product: A*B ~= Ah*Bh + Ah*Bl + Al*Bh  (all f32-accumulated WMMA)
__device__ __forceinline__ v8f wmma3(bf16x16 ah, bf16x16 al, bf16x16 bh, bf16x16 bl, v8f c) {
    c = __builtin_amdgcn_wmma_f32_16x16x32_bf16(false, ah, false, bh, (short)0, c, false, false);
    c = __builtin_amdgcn_wmma_f32_16x16x32_bf16(false, ah, false, bl, (short)0, c, false, false);
    c = __builtin_amdgcn_wmma_f32_16x16x32_bf16(false, al, false, bh, (short)0, c, false, false);
    return c;
}

__device__ __forceinline__ float clip_rsqrt_deg(int d) {
    return rsqrtf((float)(d < 1 ? 1 : d));
}

// -------- Pack a [K][N] f32 weight matrix into per-lane WMMA B-fragment layout --------
// dword index i = (ks*N + n)*16 + h*8 + p  holds bf16 pair (W[k0][n], W[k0+1][n]),
// k0 = ks*32 + h*8 + (p&3)*2 + (p>>2)*16  (ISA 7.12.2: 16-bit B 32x16, lane = n-col, half h = K-block).
__global__ void pack_weights_kernel(const float* __restrict__ W, int K, int N,
                                    unsigned* __restrict__ hi, unsigned* __restrict__ lo) {
    int total = (K * N) >> 1;
    for (int i = blockIdx.x * blockDim.x + threadIdx.x; i < total; i += gridDim.x * blockDim.x) {
        int ks  = i / (N * 16);
        int rem = i - ks * (N * 16);
        int n   = rem >> 4;
        int q   = rem & 15;
        int h   = q >> 3, p = q & 7;
        int k0  = ks * 32 + h * 8 + (p & 3) * 2 + ((p >> 2) << 4);
        float x0 = W[(size_t)k0 * N + n];
        float x1 = W[(size_t)(k0 + 1) * N + n];
        unsigned uh, ul;
        split2(x0, x1, uh, ul);
        hi[i] = uh;
        lo[i] = ul;
    }
}

// -------- Degree counting --------
__global__ void degree_kernel(const int* __restrict__ src, const int* __restrict__ dst, int E,
                              int* __restrict__ outdeg, int* __restrict__ indeg) {
    int e = blockIdx.x * blockDim.x + threadIdx.x;
    if (e < E) {
        atomicAdd(&outdeg[src[e]], 1);
        atomicAdd(&indeg[dst[e]], 1);
    }
}

// -------- Edge scatter: one wave32 per edge, 8 channels/lane --------
__global__ void scatter_kernel(const float* __restrict__ h, const int* __restrict__ src,
                               const int* __restrict__ dst, float* __restrict__ agg, int E) {
    int gid = blockIdx.x * blockDim.x + threadIdx.x;
    int e = gid >> 5;
    if (e >= E) return;
    int lane = gid & 31;
    const float4* hp = (const float4*)(h + (size_t)src[e] * H_DIM);
    float4 a = hp[lane * 2];
    float4 b = hp[lane * 2 + 1];
    float* base = agg + (size_t)dst[e] * H_DIM + lane * 8;
    atomicAdd(base + 0, a.x); atomicAdd(base + 1, a.y);
    atomicAdd(base + 2, a.z); atomicAdd(base + 3, a.w);
    atomicAdd(base + 4, b.x); atomicAdd(base + 5, b.y);
    atomicAdd(base + 6, b.z); atomicAdd(base + 7, b.w);
}

// Stage 16 rows of A (optionally row-scaled) into LDS as packed hi/lo A-fragments.
// dword i = ks*256 + lane*8 + p holds pair for (row=lane&15, h=lane>>4),
// k = ks*32 + h*8 + (p&3)*2 + (p>>2)*16  (ISA 7.12.2: 16-bit A 16x32 layout).
__device__ __forceinline__ void stage_A(const float* __restrict__ A, int row0, int K,
                                        const int* __restrict__ rowdeg, // null -> no scale
                                        unsigned* sAhi, unsigned* sAlo,
                                        int tid, int nthreads) {
    int total = (K / 32) * 256;
    for (int i = tid; i < total; i += nthreads) {
        int ks = i >> 8;
        int rem = i & 255;
        int lane = rem >> 3, p = rem & 7;
        int row = lane & 15, hh = lane >> 4;
        int k = ks * 32 + hh * 8 + (p & 3) * 2 + ((p >> 2) << 4);
        const float* ap = A + (size_t)(row0 + row) * K + k;
        float rs = rowdeg ? clip_rsqrt_deg(rowdeg[row0 + row]) : 1.0f;
        unsigned uh, ul;
        split2(ap[0] * rs, ap[1] * rs, uh, ul);
        sAhi[i] = uh;
        sAlo[i] = ul;
    }
}

// -------- GEMM1: hs0 = (feats @ W1 + bias1) * rsqrt(outdeg0)  [N0 x 256] --------
__global__ __launch_bounds__(512) void gemm1_kernel(
        const float* __restrict__ A, const unsigned* __restrict__ Bhi,
        const unsigned* __restrict__ Blo, const float* __restrict__ bias,
        const int* __restrict__ outdeg0, float* __restrict__ out) {
    __shared__ unsigned sAhi[16 * 256];
    __shared__ unsigned sAlo[16 * 256];
    __shared__ float tile[16][264];
    const int K = IN_DIM, N = H_DIM;
    int tid = threadIdx.x;
    int row0 = blockIdx.x * 16;

    stage_A(A, row0, K, nullptr, sAhi, sAlo, tid, 512);
    __syncthreads();

    int wave = tid >> 5, lane = tid & 31;
    int n0 = wave * 16;
    int col = n0 + (lane & 15);
    int h = lane >> 4;
    v8f acc = {};
    for (int ks = 0; ks < K / 32; ++ks) {
        bf16x16 ah = ld_frag(&sAhi[ks * 256 + lane * 8]);
        bf16x16 al = ld_frag(&sAlo[ks * 256 + lane * 8]);
        size_t boff = (size_t)(ks * N + col) * 16 + h * 8;
        bf16x16 bh = ld_frag(&Bhi[boff]);
        bf16x16 bl = ld_frag(&Blo[boff]);
        acc = wmma3(ah, al, bh, bl, acc);
    }
#pragma unroll
    for (int p = 0; p < 8; ++p) tile[p + h * 8][col] = acc[p];
    __syncthreads();

    for (int i = tid; i < 16 * N; i += 512) {
        int r = i >> 8, c = i & (N - 1);
        float rs = clip_rsqrt_deg(outdeg0[row0 + r]);
        out[(size_t)(row0 + r) * N + c] = (tile[r][c] + bias[c]) * rs;
    }
}

// -------- GEMM2: hs1 = LN(relu(agg0*rsqrt(indeg0) @ Wc + bc)) * rsqrt(outdeg1) --------
__global__ __launch_bounds__(512) void gemm2_kernel(
        const float* __restrict__ A, const int* __restrict__ indeg0,
        const unsigned* __restrict__ Bhi, const unsigned* __restrict__ Blo,
        const float* __restrict__ bc, const float* __restrict__ ln_g,
        const float* __restrict__ ln_b, const int* __restrict__ outdeg1,
        float* __restrict__ out) {
    __shared__ unsigned sAhi[8 * 256];
    __shared__ unsigned sAlo[8 * 256];
    __shared__ float tile[16][264];
    const int K = H_DIM, N = H_DIM;
    int tid = threadIdx.x;
    int row0 = blockIdx.x * 16;

    stage_A(A, row0, K, indeg0, sAhi, sAlo, tid, 512);
    __syncthreads();

    int wave = tid >> 5, lane = tid & 31;
    int n0 = wave * 16;
    int col = n0 + (lane & 15);
    int h = lane >> 4;
    v8f acc = {};
    for (int ks = 0; ks < K / 32; ++ks) {
        bf16x16 ah = ld_frag(&sAhi[ks * 256 + lane * 8]);
        bf16x16 al = ld_frag(&sAlo[ks * 256 + lane * 8]);
        size_t boff = (size_t)(ks * N + col) * 16 + h * 8;
        bf16x16 bh = ld_frag(&Bhi[boff]);
        bf16x16 bl = ld_frag(&Blo[boff]);
        acc = wmma3(ah, al, bh, bl, acc);
    }
    float biasc = bc[col];
#pragma unroll
    for (int p = 0; p < 8; ++p) tile[p + h * 8][col] = fmaxf(acc[p] + biasc, 0.0f);
    __syncthreads();

    // LayerNorm over H=256 per row; wave w owns row w (16 waves, 16 rows).
    int w = wave;
    float vals[8];
    float s = 0.0f, ss = 0.0f;
#pragma unroll
    for (int i = 0; i < 8; ++i) {
        float x = tile[w][lane * 8 + i];
        vals[i] = x;
        s += x;
        ss += x * x;
    }
#pragma unroll
    for (int off = 16; off > 0; off >>= 1) {
        s += __shfl_xor(s, off, 32);
        ss += __shfl_xor(ss, off, 32);
    }
    float mu = s * (1.0f / 256.0f);
    float var = ss * (1.0f / 256.0f) - mu * mu;
    float rstd = rsqrtf(var + 1e-5f);
    int grow = row0 + w;
    float rowscale = clip_rsqrt_deg(outdeg1[grow]);
#pragma unroll
    for (int i = 0; i < 8; ++i) {
        int c = lane * 8 + i;
        out[(size_t)grow * 256 + c] = ((vals[i] - mu) * rstd * ln_g[c] + ln_b[c]) * rowscale;
    }
}

// -------- GEMM3: out = rownorm(agg1*rsqrt(indeg1) @ Wo + bo)  [N2 x 128] --------
__global__ __launch_bounds__(256) void gemm3_kernel(
        const float* __restrict__ A, const int* __restrict__ indeg1,
        const unsigned* __restrict__ Bhi, const unsigned* __restrict__ Blo,
        const float* __restrict__ bo, float* __restrict__ out) {
    __shared__ unsigned sAhi[8 * 256];
    __shared__ unsigned sAlo[8 * 256];
    __shared__ float tile[16][136];
    const int K = H_DIM, N = OUT_DIM;
    int tid = threadIdx.x;
    int row0 = blockIdx.x * 16;

    stage_A(A, row0, K, indeg1, sAhi, sAlo, tid, 256);
    __syncthreads();

    int wave = tid >> 5, lane = tid & 31;   // 8 waves, N-tiles of 16 -> N=128
    int n0 = wave * 16;
    int col = n0 + (lane & 15);
    int h = lane >> 4;
    v8f acc = {};
    for (int ks = 0; ks < K / 32; ++ks) {
        bf16x16 ah = ld_frag(&sAhi[ks * 256 + lane * 8]);
        bf16x16 al = ld_frag(&sAlo[ks * 256 + lane * 8]);
        size_t boff = (size_t)(ks * N + col) * 16 + h * 8;
        bf16x16 bh = ld_frag(&Bhi[boff]);
        bf16x16 bl = ld_frag(&Blo[boff]);
        acc = wmma3(ah, al, bh, bl, acc);
    }
    float biasc = bo[col];
#pragma unroll
    for (int p = 0; p < 8; ++p) tile[p + h * 8][col] = acc[p] + biasc;
    __syncthreads();

    // L2 row-normalize: 8 waves cover 16 rows (rows wave, wave+8); 4 cols/lane.
    for (int rr = wave; rr < 16; rr += 8) {
        float vals[4];
        float ss = 0.0f;
#pragma unroll
        for (int i = 0; i < 4; ++i) {
            float x = tile[rr][lane * 4 + i];
            vals[i] = x;
            ss += x * x;
        }
#pragma unroll
        for (int off = 16; off > 0; off >>= 1) ss += __shfl_xor(ss, off, 32);
        float nrm = sqrtf(ss);
        float sc = 1.0f / fmaxf(nrm, 1e-12f);
        int grow = row0 + rr;
#pragma unroll
        for (int i = 0; i < 4; ++i)
            out[(size_t)grow * N + lane * 4 + i] = vals[i] * sc;
    }
}

extern "C" void kernel_launch(void* const* d_in, const int* in_sizes, int n_in,
                              void* d_out, int out_size, void* d_ws, size_t ws_size,
                              hipStream_t stream) {
    const float* feats = (const float*)d_in[0];
    const int* b0_src  = (const int*)d_in[1];
    const int* b0_dst  = (const int*)d_in[2];
    const int* b1_src  = (const int*)d_in[3];
    const int* b1_dst  = (const int*)d_in[4];
    const float* W1    = (const float*)d_in[5];
    const float* bias1 = (const float*)d_in[6];
    const float* Wc    = (const float*)d_in[7];
    const float* bcv   = (const float*)d_in[8];
    const float* ln_g  = (const float*)d_in[9];
    const float* ln_b  = (const float*)d_in[10];
    const float* Wo    = (const float*)d_in[11];
    const float* bov   = (const float*)d_in[12];
    const int E0 = in_sizes[1];
    const int E1 = in_sizes[3];

    // Carve workspace (all offsets 256B-aligned, deterministic).
    char* ws = (char*)d_ws;
    size_t off = 0;
    auto carve = [&](size_t bytes) -> void* {
        void* p = (void*)(ws + off);
        off += (bytes + 255) & ~(size_t)255;
        return p;
    };
    float* hs0      = (float*)carve((size_t)NN0 * H_DIM * 4);
    float* agg0     = (float*)carve((size_t)NN1 * H_DIM * 4);
    float* hs1      = (float*)carve((size_t)NN1 * H_DIM * 4);
    float* agg1     = (float*)carve((size_t)NN2 * H_DIM * 4);
    int* outdeg0    = (int*)carve((size_t)NN0 * 4);
    int* indeg0     = (int*)carve((size_t)NN1 * 4);
    int* outdeg1    = (int*)carve((size_t)NN1 * 4);
    int* indeg1     = (int*)carve((size_t)NN2 * 4);
    unsigned* W1hi  = (unsigned*)carve((size_t)IN_DIM * H_DIM / 2 * 4);
    unsigned* W1lo  = (unsigned*)carve((size_t)IN_DIM * H_DIM / 2 * 4);
    unsigned* Wchi  = (unsigned*)carve((size_t)H_DIM * H_DIM / 2 * 4);
    unsigned* Wclo  = (unsigned*)carve((size_t)H_DIM * H_DIM / 2 * 4);
    unsigned* Wohi  = (unsigned*)carve((size_t)H_DIM * OUT_DIM / 2 * 4);
    unsigned* Wolo  = (unsigned*)carve((size_t)H_DIM * OUT_DIM / 2 * 4);

    // Zero accumulators / degree counters (capture-safe stream memsets).
    hipMemsetAsync(agg0, 0, (size_t)NN1 * H_DIM * 4, stream);
    hipMemsetAsync(agg1, 0, (size_t)NN2 * H_DIM * 4, stream);
    hipMemsetAsync(outdeg0, 0, (size_t)NN0 * 4, stream);
    hipMemsetAsync(indeg0, 0, (size_t)NN1 * 4, stream);
    hipMemsetAsync(outdeg1, 0, (size_t)NN1 * 4, stream);
    hipMemsetAsync(indeg1, 0, (size_t)NN2 * 4, stream);

    degree_kernel<<<(E0 + 255) / 256, 256, 0, stream>>>(b0_src, b0_dst, E0, outdeg0, indeg0);
    degree_kernel<<<(E1 + 255) / 256, 256, 0, stream>>>(b1_src, b1_dst, E1, outdeg1, indeg1);

    pack_weights_kernel<<<256, 256, 0, stream>>>(W1, IN_DIM, H_DIM, W1hi, W1lo);
    pack_weights_kernel<<<128, 256, 0, stream>>>(Wc, H_DIM, H_DIM, Wchi, Wclo);
    pack_weights_kernel<<<64, 256, 0, stream>>>(Wo, H_DIM, OUT_DIM, Wohi, Wolo);

    gemm1_kernel<<<NN0 / 16, 512, 0, stream>>>(feats, W1hi, W1lo, bias1, outdeg0, hs0);
    scatter_kernel<<<(E0 * 32 + 255) / 256, 256, 0, stream>>>(hs0, b0_src, b0_dst, agg0, E0);
    gemm2_kernel<<<NN1 / 16, 512, 0, stream>>>(agg0, indeg0, Wchi, Wclo, bcv, ln_g, ln_b,
                                               outdeg1, hs1);
    scatter_kernel<<<(E1 * 32 + 255) / 256, 256, 0, stream>>>(hs1, b1_src, b1_dst, agg1, E1);
    gemm3_kernel<<<NN2 / 16, 256, 0, stream>>>(agg1, indeg1, Wohi, Wolo, bov, (float*)d_out);
}
